// LWP_89378269430399
// MI455X (gfx1250) — matrix operand order repaired and checked by reading, not compiled, not run on hardware
//
#include <hip/hip_runtime.h>
#include <stdint.h>

typedef _Float16 h16;
typedef __attribute__((ext_vector_type(8)))  _Float16 v8h;
typedef __attribute__((ext_vector_type(16))) _Float16 v16h;
typedef __attribute__((ext_vector_type(8)))  float    v8f;

// Build h0 = concat(x, diff, rec_prev, hidden_prev) as f16 [N, 256]
__global__ void build_h0_kernel(const float* __restrict__ x, const float* __restrict__ diff,
                                const float* __restrict__ rec, const float* __restrict__ hid,
                                h16* __restrict__ H, int n_nodes) {
  long long idx = (long long)blockIdx.x * blockDim.x + threadIdx.x;
  long long total = (long long)n_nodes * 256;
  if (idx >= total) return;
  int n = (int)(idx >> 8);
  int f = (int)(idx & 255);
  const float* s = (f < 64) ? x : (f < 128) ? diff : (f < 192) ? rec : hid;
  H[idx] = (h16)s[(long long)n * 64 + (f & 63)];
}

// In-degree counts (fp32 so the divide path is uniform)
__global__ void count_kernel(const long long* __restrict__ dst, float* __restrict__ cnt,
                             int n_edges) {
  int e = blockIdx.x * blockDim.x + threadIdx.x;
  if (e >= n_edges) return;
  atomicAdd(&cnt[(int)dst[e]], 1.0f);
}

// Edge scatter: S[dst] += h[src]; one thread handles an 8-feature chunk (16B f16 load)
__global__ void scatter_kernel(const h16* __restrict__ H, const long long* __restrict__ src,
                               const long long* __restrict__ dst, float* __restrict__ S,
                               int n_edges, int din) {
  int e = blockIdx.x * blockDim.x + threadIdx.x;
  if (e >= n_edges) return;
  int c = blockIdx.y;  // chunk of 8 features
  long long se = src[e], de = dst[e];
  v8h hv = *(const v8h*)(H + se * (long long)din + c * 8);
  float* sp = S + de * (long long)din + c * 8;
#pragma unroll
  for (int j = 0; j < 8; ++j) atomicAdd(&sp[j], (float)hv[j]);
}

// A = [mean | h] as f16 [N, 2*din]
__global__ void build_a_kernel(const float* __restrict__ S, const float* __restrict__ cnt,
                               const h16* __restrict__ H, h16* __restrict__ A,
                               int n_nodes, int din, int shift) {
  long long idx = (long long)blockIdx.x * blockDim.x + threadIdx.x;
  long long total = (long long)n_nodes * din;
  if (idx >= total) return;
  int n = (int)(idx >> shift);
  int f = (int)(idx & (din - 1));
  float c = cnt[n];
  c = (c < 1.0f) ? 1.0f : c;
  long long base = (long long)n * (din << 1);
  A[base + f]       = (h16)(S[idx] / c);
  A[base + din + f] = H[idx];
}

// Wt[j][k] = (k < din ? Wl[k][j] : Wr[k-din][j]) as f16, row-major over K (B pre-transpose)
__global__ void conv_w_kernel(const float* __restrict__ Wl, const float* __restrict__ Wr,
                              h16* __restrict__ Wt, int din, int dout) {
  int idx = blockIdx.x * blockDim.x + threadIdx.x;
  int K = din * 2;
  if (idx >= dout * K) return;
  int j = idx / K;
  int k = idx - j * K;
  float v = (k < din) ? Wl[(long long)k * dout + j] : Wr[(long long)(k - din) * dout + j];
  Wt[(long long)j * K + k] = (h16)v;
}

// WMMA GEMM: Hout[n, j] = relu(A[n,:] . Wt[j,:] + bias[j]); dout must be 128.
// One wave -> one 16x16 tile; block of 8 waves covers all 8 column tiles.
__global__ void gemm_wmma_kernel(const h16* __restrict__ A, const h16* __restrict__ Wt,
                                 const float* __restrict__ bias, h16* __restrict__ Hout,
                                 int n_nodes, int K, int dout, int do_relu) {
  const int lane  = threadIdx.x & 31;
  const int wave  = threadIdx.x >> 5;   // column tile 0..7
  const int mtile = blockIdx.x;
  const int mrow  = lane & 15;
  const int g     = lane >> 4;          // lane-group (K split per ISA layout)
  const h16* arow = A  + (long long)(mtile * 16 + mrow) * K;
  const h16* brow = Wt + (long long)(wave  * 16 + mrow) * K;
  v8f acc = {};
  for (int kk = 0; kk < K; kk += 32) {
    // A 16x32 f16 fragment: halves 0..7 = K in [g*8, g*8+8), halves 8..15 = +16
    v8h a0 = *(const v8h*)(arow + kk + g * 8);
    v8h a1 = *(const v8h*)(arow + kk + g * 8 + 16);
    // B 32x16 f16 fragment: halves 0..7 = K in [g*8, g*8+8), halves 8..15 = +16
    v8h b0 = *(const v8h*)(brow + kk + g * 8);
    v8h b1 = *(const v8h*)(brow + kk + g * 8 + 16);
    v16h av = __builtin_shufflevector(a0, a1, 0,1,2,3,4,5,6,7,8,9,10,11,12,13,14,15);
    v16h bv = __builtin_shufflevector(b0, b1, 0,1,2,3,4,5,6,7,8,9,10,11,12,13,14,15);
    acc = __builtin_amdgcn_wmma_f32_16x16x32_f16(false, av, false, bv, (short)0, acc,
                                                 false, false);
  }
  const int col = wave * 16 + (lane & 15);
  const float bv = bias[col];
#pragma unroll
  for (int r = 0; r < 8; ++r) {
    int row = mtile * 16 + r + g * 8;   // C/D layout: VGPR r -> M = r (+8 for lanes 16..31)
    if (row < n_nodes) {
      float v = acc[r] + bv;
      if (do_relu) v = (v > 0.0f) ? v : 0.0f;
      Hout[(long long)row * dout + col] = (h16)v;
    }
  }
}

// Last layer (dout=1): dot + bias -> sigmoid -> d_out[n]; atomic mean into d_out[N]
__global__ void final_kernel(const h16* __restrict__ A, const h16* __restrict__ w,
                             const float* __restrict__ bias, float* __restrict__ out,
                             int n_nodes, int K) {
  int n = blockIdx.x * blockDim.x + threadIdx.x;
  if (n >= n_nodes) return;
  const h16* ap = A + (long long)n * K;
  float s = bias[0];
  for (int k = 0; k < K; k += 8) {
    v8h av = *(const v8h*)(ap + k);
    v8h wv = *(const v8h*)(w + k);
#pragma unroll
    for (int j = 0; j < 8; ++j) s += (float)av[j] * (float)wv[j];
  }
  float o = 1.0f / (1.0f + __expf(-s));
  out[n] = o;
  atomicAdd(&out[n_nodes], o * (1.0f / (float)n_nodes));
}

extern "C" void kernel_launch(void* const* d_in, const int* in_sizes, int n_in,
                              void* d_out, int out_size, void* d_ws, size_t ws_size,
                              hipStream_t stream) {
  const float* x    = (const float*)d_in[0];
  const float* diff = (const float*)d_in[1];
  const float* rec  = (const float*)d_in[2];
  const float* hid  = (const float*)d_in[3];
  const long long* ei = (const long long*)d_in[4];  // int64 [2, E]
  const float* Wl[4] = {(const float*)d_in[5], (const float*)d_in[8],
                        (const float*)d_in[11], (const float*)d_in[14]};
  const float* Wr[4] = {(const float*)d_in[6], (const float*)d_in[9],
                        (const float*)d_in[12], (const float*)d_in[15]};
  const float* bs[4] = {(const float*)d_in[7], (const float*)d_in[10],
                        (const float*)d_in[13], (const float*)d_in[16]};
  const int N = in_sizes[0] / 64;
  const int E = in_sizes[4] / 2;
  const long long* srcv = ei;
  const long long* dstv = ei + E;
  const int Npad = (N + 15) & ~15;

  // Carve workspace (256B aligned regions)
  char* p = (char*)d_ws;
  auto carve = [&](size_t bytes) -> char* {
    char* r = p;
    p += (bytes + 255) & ~(size_t)255;
    return r;
  };
  h16*   H   = (h16*)  carve((size_t)Npad * 256 * sizeof(h16));   // activations (<=256 feats)
  h16*   Am  = (h16*)  carve((size_t)Npad * 512 * sizeof(h16));   // [mean|h] (<=512 cols)
  float* S   = (float*)carve((size_t)N * 256 * sizeof(float));    // fp32 segment sums
  float* cnt = (float*)carve((size_t)N * sizeof(float));          // in-degrees
  h16*   Wt  = (h16*)  carve((size_t)512 * 128 * sizeof(h16));    // transposed f16 weights
  float* out = (float*)d_out;

  // Initial features + degree counts (counts are layer-invariant)
  {
    long long tot = (long long)N * 256;
    build_h0_kernel<<<(unsigned)((tot + 255) / 256), 256, 0, stream>>>(x, diff, rec, hid, H, N);
    hipMemsetAsync(cnt, 0, (size_t)N * sizeof(float), stream);
    count_kernel<<<(E + 255) / 256, 256, 0, stream>>>(dstv, cnt, E);
  }

  const int dins[4]   = {256, 128, 128, 128};
  const int shifts[4] = {8, 7, 7, 7};
  for (int l = 0; l < 4; ++l) {
    const int din  = dins[l];
    const int K    = 2 * din;
    const int dout = (l == 3) ? 1 : 128;

    hipMemsetAsync(S, 0, (size_t)N * din * sizeof(float), stream);
    dim3 sg((unsigned)((E + 255) / 256), (unsigned)(din / 8));
    scatter_kernel<<<sg, 256, 0, stream>>>(H, srcv, dstv, S, E, din);

    long long tot = (long long)N * din;
    build_a_kernel<<<(unsigned)((tot + 255) / 256), 256, 0, stream>>>(S, cnt, H, Am, N, din,
                                                                      shifts[l]);
    int wtot = dout * K;
    conv_w_kernel<<<(wtot + 255) / 256, 256, 0, stream>>>(Wl[l], Wr[l], Wt, din, dout);

    if (l < 3) {
      gemm_wmma_kernel<<<Npad / 16, 256, 0, stream>>>(Am, Wt, bs[l], H, N, K, dout, 1);
    } else {
      hipMemsetAsync(out + N, 0, sizeof(float), stream);
      final_kernel<<<(N + 255) / 256, 256, 0, stream>>>(Am, Wt, bs[l], out, N, K);
    }
  }
}
